// NetTGCNTwoLayer_13288628814251
// MI455X (gfx1250) — compile-verified, matrix operand order checked by reading, not run on hardware
//
#include <hip/hip_runtime.h>
#include <hip/hip_bf16.h>

// ---------------------------------------------------------------------------
// NetTGCN two-layer ChebConv for MI455X (gfx1250, wave32, WMMA).
// Dominant op: b_map (4096x8192 f32) x h1 (8192x256) GEMM -> WMMA bf16 path.
// A is converted f32->bf16 in the LDS fill (one-pass 128MB HBM read); B is
// pre-swizzled into the WMMA fragment layout so both operand fragments load
// as ds_load_b128 pairs (no ds_load_u16 storm). Everything else is L2-bound.
// ---------------------------------------------------------------------------

typedef __attribute__((ext_vector_type(16))) __bf16 v16bf;
typedef __attribute__((ext_vector_type(8)))  float  v8f;

static constexpr int BSZ = 4;
static constexpr int N1  = 8192;
static constexpr int N2  = 4096;
static constexpr int TT  = 15;
static constexpr int E1  = 131072;
static constexpr int E2  = 65536;
static constexpr int KCH = 12;
static constexpr int G1  = 64;
static constexpr int G2  = 32;
static constexpr int NC  = 6;

// ---------------------------------------------------------------- utilities
__global__ void k_zero(float* __restrict__ p, size_t n) {
  size_t i = (size_t)blockIdx.x * blockDim.x + threadIdx.x;
  if (i < n) p[i] = 0.0f;
}

__global__ void k_deg(const int* __restrict__ ei, const float* __restrict__ w,
                      float* __restrict__ deg, int E) {
  int e = blockIdx.x * blockDim.x + threadIdx.x;
  if (e < E) atomicAdd(&deg[ei[E + e]], w[e]);
}

__global__ void k_norm(const int* __restrict__ ei, const float* __restrict__ w,
                       const float* __restrict__ deg, float* __restrict__ norm, int E) {
  int e = blockIdx.x * blockDim.x + threadIdx.x;
  if (e >= E) return;
  float ds = deg[ei[e]], dt = deg[ei[E + e]];
  float is = ds > 0.0f ? rsqrtf(ds) : 0.0f;
  float it = dt > 0.0f ? rsqrtf(dt) : 0.0f;
  norm[e] = -w[e] * is * it;
}

// x:(B,N1,T) -> h0:(N1,T,B) row layout [t*4+b]
__global__ void k_permute_x(const float* __restrict__ x, float* __restrict__ h0) {
  size_t gid = (size_t)blockIdx.x * blockDim.x + threadIdx.x;
  if (gid >= (size_t)N1 * TT * BSZ) return;
  int n = (int)(gid / (TT * BSZ));
  int r = (int)(gid % (TT * BSZ));
  int t = r >> 2, b = r & 3;
  h0[gid] = x[(size_t)b * N1 * TT + (size_t)n * TT + t];
}

// out[dst] += norm[e] * h[src]  (rows of `rowlen` floats), f32 atomics (L2)
__global__ void k_prop(const int* __restrict__ ei, const float* __restrict__ norm,
                       const float* __restrict__ h, float* __restrict__ out,
                       int E, int rowlen) {
  size_t gid = (size_t)blockIdx.x * blockDim.x + threadIdx.x;
  size_t total = (size_t)E * rowlen;
  if (gid >= total) return;
  int e = (int)(gid / rowlen);
  int d = (int)(gid % rowlen);
  int s = ei[e], t = ei[E + e];
  atomicAdd(&out[(size_t)t * rowlen + d], norm[e] * h[(size_t)s * rowlen + d]);
}

// p = 2*p - prev   (Chebyshev recursion combine)
__global__ void k_combine(float* __restrict__ p, const float* __restrict__ prev, size_t n) {
  size_t i = (size_t)blockIdx.x * blockDim.x + threadIdx.x;
  if (i < n) p[i] = 2.0f * p[i] - prev[i];
}

// y[n,g,b] += sum_d T[n,d,b] * W[d,g]   (B=4 fixed)
__global__ void k_einsum(const float* __restrict__ T, const float* __restrict__ W,
                         float* __restrict__ y, int N, int D, int G) {
  size_t gid = (size_t)blockIdx.x * blockDim.x + threadIdx.x;
  if (gid >= (size_t)N * G * BSZ) return;
  int n = (int)(gid / (G * BSZ));
  int r = (int)(gid % (G * BSZ));
  int g = r >> 2, b = r & 3;
  const float* tr = T + (size_t)n * D * BSZ + b;
  float acc = 0.0f;
  for (int d = 0; d < D; ++d) acc = fmaf(tr[d * BSZ], W[d * G + g], acc);
  y[gid] += acc;
}

__global__ void k_bias_act(float* __restrict__ y, const float* __restrict__ bias,
                           int N, int G, int relu) {
  size_t gid = (size_t)blockIdx.x * blockDim.x + threadIdx.x;
  if (gid >= (size_t)N * G * BSZ) return;
  int g = ((int)(gid % (G * BSZ))) >> 2;
  float v = y[gid] + bias[g];
  y[gid] = relu ? fmaxf(v, 0.0f) : v;
}

// f32 (8192x256 row-major) -> bf16 pre-swizzled to the WMMA B-fragment layout:
// dst[kt*8192 + n*32 + slot] = y[(kt*32 + perm(slot))*256 + n]
// perm swaps K-blocks [8..15] <-> [16..23] (involution), matching the 16-bit
// operand lane/K mapping so fragments are 32B-contiguous in LDS.
__global__ void k_pack_bf16_sw(const float* __restrict__ y, __bf16* __restrict__ dst,
                               size_t n) {
  size_t gid = (size_t)blockIdx.x * blockDim.x + threadIdx.x;
  if (gid >= n) return;
  int kt   = (int)(gid >> 13);          // / 8192
  int r    = (int)(gid & 8191);
  int col  = r >> 5;
  int slot = r & 31;
  int kb = slot & 24;
  int k  = (kb == 8) ? slot + 8 : (kb == 16) ? slot - 8 : slot;
  dst[gid] = (__bf16)y[((size_t)kt * 32 + k) * 256 + col];
}

// ---------------------------------------------------------------- WMMA GEMM
// C(4096x256) = A(4096x8192, f32 -> bf16 in LDS) x B(8192x256, bf16 swizzled)
// Grid: 128 WGs, tile 32x256, K-slab 64 (2 WMMA k-tiles per barrier phase).
// 8 waves: wm in {0,1} (16 rows), wn in {0..3} (64 cols): 8 wmma / slab / wave.
__global__ __launch_bounds__(256) void k_gemm_bmap(const float* __restrict__ A,
                                                   const __bf16* __restrict__ BmSw,
                                                   float* __restrict__ C) {
  constexpr int KK = 8192, NN = 256, TM = 32, KS = 64;
  __shared__ __bf16 As[TM * KS];        // [m][k]            4 KB
  __shared__ __bf16 Bs[2 * NN * 32];    // [kt][n][slot]    32 KB
  const int m0   = blockIdx.x * TM;
  const int tid  = threadIdx.x;
  const int lane = tid & 31;
  const int wave = tid >> 5;
  const int wm   = wave & 1;            // 0..1
  const int wn   = wave >> 1;           // 0..3
  const int lg   = lane >> 4;
  const int ml   = lane & 15;
  v8f acc[4] = {};

  for (int kc = 0; kc < KK; kc += KS) {
    __syncthreads();
    {   // A tile: 32x64 f32 -> bf16 (8 elems / thread, 2x float4 loads)
      int idx = tid * 8;
      int r = idx >> 6, c = idx & 63;
      const float* ap = A + (size_t)(m0 + r) * KK + kc + c;
      float4 f0 = ((const float4*)ap)[0];
      float4 f1 = ((const float4*)ap)[1];
      __bf16* d = &As[idx];
      d[0] = (__bf16)f0.x; d[1] = (__bf16)f0.y; d[2] = (__bf16)f0.z; d[3] = (__bf16)f0.w;
      d[4] = (__bf16)f1.x; d[5] = (__bf16)f1.y; d[6] = (__bf16)f1.z; d[7] = (__bf16)f1.w;
      if (kc + KS < KK) __builtin_prefetch(ap + KS, 0, 1);   // global_prefetch_b8
    }
    {   // B slab: 2 swizzled 32-k groups, layout-preserving copy (8x uint4)
      const uint4* s = (const uint4*)(BmSw + (size_t)(kc >> 5) * (NN * 32) + tid * 64);
      uint4* d = (uint4*)(&Bs[tid * 64]);
      #pragma unroll
      for (int q = 0; q < 8; ++q) d[q] = s[q];
    }
    __syncthreads();

    #pragma unroll
    for (int t = 0; t < 2; ++t) {
      // A fragment: two contiguous 8-elem K runs -> 2x ds_load_b128
      v16bf af;
      #pragma unroll
      for (int i = 0; i < 16; ++i) {
        int k = i + ((i >= 8) ? 8 : 0) + 8 * lg;
        af[i] = As[(wm * 16 + ml) * KS + t * 32 + k];
      }
      #pragma unroll
      for (int j = 0; j < 4; ++j) {
        // B fragment: 32B contiguous in swizzled LDS -> 2x ds_load_b128
        int ncol = wn * 64 + j * 16 + ml;
        v16bf bfr = *(const v16bf*)(&Bs[t * (NN * 32) + ncol * 32 + lg * 16]);
        acc[j] = __builtin_amdgcn_wmma_f32_16x16x32_bf16(
            false, af, false, bfr, (short)0, acc[j], false, false);
      }
    }
  }

  // C/D f32 layout: element (m,n): vgpr = m%8, lane = (n&15) + 16*(m/8)
  #pragma unroll
  for (int j = 0; j < 4; ++j) {
    #pragma unroll
    for (int r = 0; r < 8; ++r) {
      int m = m0 + wm * 16 + r + 8 * lg;
      int n = wn * 64 + j * 16 + ml;
      C[(size_t)m * NN + n] = acc[j][r];
    }
  }
}

// ---------------------------------------------------------------- FC + LSM
// logits[r,c] = dot(y2_linear[r*131072 : +131072], fc_w[c]) + fc_b[c]
__global__ __launch_bounds__(256) void k_fc(const float* __restrict__ flat,
                                            const float* __restrict__ fcw,
                                            const float* __restrict__ fcb,
                                            float* __restrict__ logits, int len) {
  int r = blockIdx.x / NC, c = blockIdx.x % NC;
  const float* a = flat + (size_t)r * len;
  const float* w = fcw + (size_t)c * len;
  float s = 0.0f;
  for (int j = threadIdx.x; j < len; j += 256) s = fmaf(a[j], w[j], s);
  __shared__ float red[256];
  red[threadIdx.x] = s;
  __syncthreads();
  for (int off = 128; off > 0; off >>= 1) {
    if (threadIdx.x < off) red[threadIdx.x] += red[threadIdx.x + off];
    __syncthreads();
  }
  if (threadIdx.x == 0) logits[blockIdx.x] = red[0] + fcb[c];
}

__global__ void k_log_softmax(const float* __restrict__ logits, float* __restrict__ out) {
  int r = threadIdx.x;
  if (r >= BSZ) return;
  float m = -3.4e38f;
  for (int c = 0; c < NC; ++c) m = fmaxf(m, logits[r * NC + c]);
  float s = 0.0f;
  for (int c = 0; c < NC; ++c) s += expf(logits[r * NC + c] - m);
  float lse = m + logf(s);
  for (int c = 0; c < NC; ++c) out[r * NC + c] = logits[r * NC + c] - lse;
}

// ---------------------------------------------------------------- launcher
static inline unsigned blks(size_t n, unsigned b) { return (unsigned)((n + b - 1) / b); }

extern "C" void kernel_launch(void* const* d_in, const int* in_sizes, int n_in,
                              void* d_out, int out_size, void* d_ws, size_t ws_size,
                              hipStream_t stream) {
  (void)in_sizes; (void)n_in; (void)out_size; (void)ws_size;
  const float* x    = (const float*)d_in[0];
  const int*   ei1  = (const int*)  d_in[1];
  const float* ew1  = (const float*)d_in[2];
  const int*   ei2  = (const int*)  d_in[3];
  const float* ew2  = (const float*)d_in[4];
  const float* bmap = (const float*)d_in[5];
  const float* W1   = (const float*)d_in[6];
  const float* b1   = (const float*)d_in[7];
  const float* W2   = (const float*)d_in[8];
  const float* b2   = (const float*)d_in[9];
  const float* fcw  = (const float*)d_in[10];
  const float* fcb  = (const float*)d_in[11];
  float* out = (float*)d_out;

  // workspace carving (float words)
  float* ws = (float*)d_ws;
  size_t off = 0;
  float* deg1  = ws + off; off += N1;
  float* norm1 = ws + off; off += E1;
  float* deg2  = ws + off; off += N2;
  float* norm2 = ws + off; off += E2;
  float* h0    = ws + off; off += (size_t)N1 * TT * BSZ;          // 491520
  float* t0    = ws + off; off += 1048576;                         // max rowset N2*256
  float* t1    = ws + off; off += 1048576;
  float* t2    = ws + off; off += 1048576;
  float* y1    = ws + off; off += (size_t)N1 * G1 * BSZ;          // 2097152
  __bf16* y1bf = (__bf16*)(ws + off); off += (size_t)N1 * G1 * BSZ / 2; // bf16 swizzled
  float* h2    = ws + off; off += (size_t)N2 * G1 * BSZ;          // 1048576
  float* y2    = ws + off; off += (size_t)N2 * G2 * BSZ;          // 524288
  float* lgt   = ws + off; off += 32;

  const int TB = 256;
  const size_t row1 = (size_t)TT * BSZ;   // 60
  const size_t row2 = (size_t)G1 * BSZ;   // 256

  // ---- graph norms
  k_zero<<<blks(N1, TB), TB, 0, stream>>>(deg1, N1);
  k_zero<<<blks(N2, TB), TB, 0, stream>>>(deg2, N2);
  k_deg<<<blks(E1, TB), TB, 0, stream>>>(ei1, ew1, deg1, E1);
  k_deg<<<blks(E2, TB), TB, 0, stream>>>(ei2, ew2, deg2, E2);
  k_norm<<<blks(E1, TB), TB, 0, stream>>>(ei1, ew1, deg1, norm1, E1);
  k_norm<<<blks(E2, TB), TB, 0, stream>>>(ei2, ew2, deg2, norm2, E2);

  // ---- layer 1 input
  k_permute_x<<<blks((size_t)N1 * row1, TB), TB, 0, stream>>>(x, h0);

  // ---- ChebConv 1: D=15, G=64, rows of 60 floats
  k_zero<<<blks((size_t)N1 * G1 * BSZ, TB), TB, 0, stream>>>(y1, (size_t)N1 * G1 * BSZ);
  k_einsum<<<blks((size_t)N1 * G1 * BSZ, TB), TB, 0, stream>>>(h0, W1, y1, N1, TT, G1);
  k_zero<<<blks((size_t)N1 * row1, TB), TB, 0, stream>>>(t0, (size_t)N1 * row1);
  k_prop<<<blks((size_t)E1 * row1, TB), TB, 0, stream>>>(ei1, norm1, h0, t0, E1, (int)row1);
  k_einsum<<<blks((size_t)N1 * G1 * BSZ, TB), TB, 0, stream>>>(t0, W1 + TT * G1, y1, N1, TT, G1);
  {
    float* prev = h0; float* cur = t0; float* nxt = t1;
    for (int k = 2; k < KCH; ++k) {
      k_zero<<<blks((size_t)N1 * row1, TB), TB, 0, stream>>>(nxt, (size_t)N1 * row1);
      k_prop<<<blks((size_t)E1 * row1, TB), TB, 0, stream>>>(ei1, norm1, cur, nxt, E1, (int)row1);
      k_combine<<<blks((size_t)N1 * row1, TB), TB, 0, stream>>>(nxt, prev, (size_t)N1 * row1);
      k_einsum<<<blks((size_t)N1 * G1 * BSZ, TB), TB, 0, stream>>>(nxt, W1 + (size_t)k * TT * G1,
                                                                   y1, N1, TT, G1);
      float* old_prev = prev; prev = cur; cur = nxt;
      nxt = (old_prev == h0) ? t2 : old_prev;
    }
  }
  k_bias_act<<<blks((size_t)N1 * G1 * BSZ, TB), TB, 0, stream>>>(y1, b1, N1, G1, 1);

  // ---- pooling GEMM (WMMA): h2 = b_map x relu(y1)
  k_pack_bf16_sw<<<blks((size_t)N1 * G1 * BSZ, TB), TB, 0, stream>>>(
      y1, y1bf, (size_t)N1 * G1 * BSZ);
  k_gemm_bmap<<<dim3(N2 / 32), 256, 0, stream>>>(bmap, y1bf, h2);

  // ---- ChebConv 2: D=64, G=32, rows of 256 floats
  k_zero<<<blks((size_t)N2 * G2 * BSZ, TB), TB, 0, stream>>>(y2, (size_t)N2 * G2 * BSZ);
  k_einsum<<<blks((size_t)N2 * G2 * BSZ, TB), TB, 0, stream>>>(h2, W2, y2, N2, G1, G2);
  k_zero<<<blks((size_t)N2 * row2, TB), TB, 0, stream>>>(t0, (size_t)N2 * row2);
  k_prop<<<blks((size_t)E2 * row2, TB), TB, 0, stream>>>(ei2, norm2, h2, t0, E2, (int)row2);
  k_einsum<<<blks((size_t)N2 * G2 * BSZ, TB), TB, 0, stream>>>(t0, W2 + G1 * G2, y2, N2, G1, G2);
  {
    float* prev = h2; float* cur = t0; float* nxt = t1;
    for (int k = 2; k < KCH; ++k) {
      k_zero<<<blks((size_t)N2 * row2, TB), TB, 0, stream>>>(nxt, (size_t)N2 * row2);
      k_prop<<<blks((size_t)E2 * row2, TB), TB, 0, stream>>>(ei2, norm2, cur, nxt, E2, (int)row2);
      k_combine<<<blks((size_t)N2 * row2, TB), TB, 0, stream>>>(nxt, prev, (size_t)N2 * row2);
      k_einsum<<<blks((size_t)N2 * G2 * BSZ, TB), TB, 0, stream>>>(nxt, W2 + (size_t)k * G1 * G2,
                                                                   y2, N2, G1, G2);
      float* old_prev = prev; prev = cur; cur = nxt;
      nxt = (old_prev == h2) ? t2 : old_prev;
    }
  }
  k_bias_act<<<blks((size_t)N2 * G2 * BSZ, TB), TB, 0, stream>>>(y2, b2, N2, G2, 0);

  // ---- FC + log_softmax  (faithful reshape: flat row r = y2 linear slice)
  k_fc<<<dim3(BSZ * NC), 256, 0, stream>>>(y2, fcw, fcb, lgt, N2 * G2 * BSZ / BSZ);
  k_log_softmax<<<1, 32, 0, stream>>>(lgt, out);
}
